// MultiHeadAttention_72327249264822
// MI455X (gfx1250) — matrix operand order repaired and checked by reading, not compiled, not run on hardware
//
#include <hip/hip_runtime.h>

// ---------------------------------------------------------------------------
// MI455X (gfx1250) multi-head attention forward.
// All matmuls: v_wmma_f32_16x16x32_bf16 (wave32, f32 accumulate).
// Flash-attention K/V staging: Tensor Data Mover (tensor_load_to_lds) with
// s_wait_tensorcnt synchronization.
// ---------------------------------------------------------------------------

typedef __bf16 bf16;
typedef __attribute__((ext_vector_type(16))) __bf16 v16bf;
typedef __attribute__((ext_vector_type(8)))  __bf16 v8bf;
typedef __attribute__((ext_vector_type(4)))  __bf16 v4bf;
typedef __attribute__((ext_vector_type(8)))  float  v8f;
typedef __attribute__((ext_vector_type(4)))  unsigned int u32x4;
typedef __attribute__((ext_vector_type(8)))  int i32x8;
typedef __attribute__((ext_vector_type(4)))  int i32x4;

#define D_MODEL 1024
#define N_HEADS 16
#define HEAD_DIM 64
#define SEQ 2048
#define BATCH 2

// Load a 16-element bf16 WMMA fragment as two 16-byte chunks.
__device__ __forceinline__ v16bf ld_frag(const bf16* p0, const bf16* p1) {
  union { v16bf v; v8bf h[2]; } u;
  u.h[0] = *(const v8bf*)p0;
  u.h[1] = *(const v8bf*)p1;
  return u.v;
}

#define WMMA_BF16(A, B, C) \
  __builtin_amdgcn_wmma_f32_16x16x32_bf16(false, (A), false, (B), (short)0, (C), false, false)

// ---------------------------------------------------------------------------
// Tensor Data Mover: stage a 2D tile (tile_w contiguous elems x tile_h rows,
// row stride stride_elems, 2-byte elements) from global into LDS at byte
// offset lds_off.  D# layout per CDNA5 ISA ch.8 (group0 + group1, 2D form).
// ---------------------------------------------------------------------------
__device__ __forceinline__ void tdm_load_2d(unsigned lds_off, const void* gaddr,
                                            unsigned tile_w, unsigned tile_h,
                                            unsigned stride_elems) {
#if __has_builtin(__builtin_amdgcn_tensor_load_to_lds)
  unsigned long long ga = (unsigned long long)gaddr;
  u32x4 g0;
  g0[0] = 1u;                                        // count=1 (valid), user mode
  g0[1] = lds_off;                                   // lds_addr (bytes)
  g0[2] = (unsigned)(ga & 0xffffffffu);              // global_addr[31:0]
  g0[3] = (unsigned)((ga >> 32) & 0x1ffffffu) | (2u << 30);  // addr[56:32] | type=2
  i32x8 g1;
  g1[0] = 0x00010000;            // workgroup_mask=0, data_size=1 (2 bytes)
  g1[1] = 0;                     // atomic_barrier_addr=0, tensor_dim0[15:0]=0
  g1[2] = 0x4000;                // tensor_dim0=0x40000000 (hi16), tensor_dim1 lo16=0
  g1[3] = (int)(0x4000u | (tile_w << 16));  // tensor_dim1 hi16, tile_dim0
  g1[4] = (int)tile_h;           // tile_dim1 ; tile_dim2=0
  g1[5] = (int)stride_elems;     // tensor_dim0_stride[31:0]
  g1[6] = 0;                     // stride hi, tensor_dim1_stride lo
  g1[7] = 0;
  i32x4 z4 = {0, 0, 0, 0};
#if defined(__clang_major__) && (__clang_major__ >= 23)
  i32x8 z8 = {0, 0, 0, 0, 0, 0, 0, 0};
  __builtin_amdgcn_tensor_load_to_lds(g0, g1, z4, z4, z8, 0);
#else
  __builtin_amdgcn_tensor_load_to_lds(g0, g1, z4, z4, 0);
#endif
#else
  // Fallback: cooperative per-wave copy (16B per lane per step).
  int lane = threadIdx.x & 31;
  bf16* dst = (bf16*)((char*)nullptr);  // unused marker; real fallback below
  (void)dst;
  bf16* ldsb;
  {
    extern __shared__ bf16 _lds_base[];
    ldsb = _lds_base;  // only used in fallback builds
  }
  bf16* d = (bf16*)((char*)ldsb + lds_off);
  const bf16* g = (const bf16*)gaddr;
  int vpr = tile_w >> 3;
  int total = (int)(tile_h * vpr);
  for (int i = lane; i < total; i += 32) {
    int r = i / vpr, c = i - r * vpr;
    *(v8bf*)(d + r * tile_w + c * 8) = *(const v8bf*)(g + (size_t)r * stride_elems + c * 8);
  }
#endif
}

__device__ __forceinline__ void tdm_wait() {
#if __has_builtin(__builtin_amdgcn_s_wait_tensorcnt)
  __builtin_amdgcn_s_wait_tensorcnt(0);
#endif
  asm volatile("" ::: "memory");
}

// ---------------------------------------------------------------------------
// fp32 -> bf16 conversion (4-wide, grid-stride)
// ---------------------------------------------------------------------------
__global__ __launch_bounds__(256) void cvt_f32_bf16(const float* __restrict__ src,
                                                    bf16* __restrict__ dst, int n4) {
  int i = blockIdx.x * blockDim.x + threadIdx.x;
  int stride = gridDim.x * blockDim.x;
  const float4* s4 = (const float4*)src;
  v4bf* d4 = (v4bf*)dst;
  for (; i < n4; i += stride) {
    float4 v = s4[i];
    v4bf o;
    o[0] = (bf16)v.x; o[1] = (bf16)v.y; o[2] = (bf16)v.z; o[3] = (bf16)v.w;
    d4[i] = o;
  }
}

// ---------------------------------------------------------------------------
// GEMM: Y[M,N] = X[M,K] @ W[N,K]^T   (bf16 in, f32 accumulate)
// One wave computes a 32x64 tile: 2 A-frags x 4 B-frags = 8 wmma per K-chunk,
// with explicit register double-buffering of the next chunk's fragments.
// mode 0: bf16 [M,N] | mode 1: bf16 vt[b][h][d][s] (V) | mode 2: f32 [M,N]
// ---------------------------------------------------------------------------
__global__ __launch_bounds__(256) void gemm_bf16_wmma(
    const bf16* __restrict__ X, const bf16* __restrict__ W,
    bf16* __restrict__ Yb, float* __restrict__ Yf,
    int M, int N, int K, int mode, int S) {
  int wave = (blockIdx.x * blockDim.x + threadIdx.x) >> 5;
  int lane = threadIdx.x & 31;
  int l = lane & 15, half = lane >> 4;
  int ntiles = N >> 6;
  int mt = wave / ntiles;
  int nt = wave - mt * ntiles;
  int m0 = mt << 5, n0 = nt << 6;
  if (m0 >= M) return;

  const bf16* arow0 = X + (size_t)(m0 + l) * K + half * 16;
  const bf16* arow1 = X + (size_t)(m0 + 16 + l) * K + half * 16;
  const bf16* brow  = W + (size_t)(n0 + l) * K + half * 16;

  v8f acc00 = {}, acc01 = {}, acc02 = {}, acc03 = {};
  v8f acc10 = {}, acc11 = {}, acc12 = {}, acc13 = {};

  // prologue: fragments for kk = 0
  v16bf a0 = ld_frag(arow0, arow0 + 8);
  v16bf a1 = ld_frag(arow1, arow1 + 8);
  v16bf w0 = ld_frag(brow,           brow + 8);
  v16bf w1 = ld_frag(brow + 16 * K,  brow + 16 * K + 8);
  v16bf w2 = ld_frag(brow + 32 * K,  brow + 32 * K + 8);
  v16bf w3 = ld_frag(brow + 48 * K,  brow + 48 * K + 8);

  for (int kk = 32; kk < K; kk += 32) {
    __builtin_prefetch(arow0 + kk + 32, 0, 0);
    __builtin_prefetch(brow + kk + 32, 0, 0);
    // issue next chunk's loads before consuming current fragments
    v16bf na0 = ld_frag(arow0 + kk, arow0 + kk + 8);
    v16bf na1 = ld_frag(arow1 + kk, arow1 + kk + 8);
    v16bf nw0 = ld_frag(brow + kk,          brow + kk + 8);
    v16bf nw1 = ld_frag(brow + 16 * K + kk, brow + 16 * K + kk + 8);
    v16bf nw2 = ld_frag(brow + 32 * K + kk, brow + 32 * K + kk + 8);
    v16bf nw3 = ld_frag(brow + 48 * K + kk, brow + 48 * K + kk + 8);

    acc00 = WMMA_BF16(a0, w0, acc00);
    acc01 = WMMA_BF16(a0, w1, acc01);
    acc02 = WMMA_BF16(a0, w2, acc02);
    acc03 = WMMA_BF16(a0, w3, acc03);
    acc10 = WMMA_BF16(a1, w0, acc10);
    acc11 = WMMA_BF16(a1, w1, acc11);
    acc12 = WMMA_BF16(a1, w2, acc12);
    acc13 = WMMA_BF16(a1, w3, acc13);

    a0 = na0; a1 = na1; w0 = nw0; w1 = nw1; w2 = nw2; w3 = nw3;
  }
  acc00 = WMMA_BF16(a0, w0, acc00);
  acc01 = WMMA_BF16(a0, w1, acc01);
  acc02 = WMMA_BF16(a0, w2, acc02);
  acc03 = WMMA_BF16(a0, w3, acc03);
  acc10 = WMMA_BF16(a1, w0, acc10);
  acc11 = WMMA_BF16(a1, w1, acc11);
  acc12 = WMMA_BF16(a1, w2, acc12);
  acc13 = WMMA_BF16(a1, w3, acc13);

  v8f acc[2][4] = {{acc00, acc01, acc02, acc03}, {acc10, acc11, acc12, acc13}};
#pragma unroll
  for (int rh = 0; rh < 2; ++rh) {
#pragma unroll
    for (int t = 0; t < 4; ++t) {
#pragma unroll
      for (int r = 0; r < 8; ++r) {
        int row = m0 + rh * 16 + r + 8 * half;
        int col = n0 + 16 * t + l;
        float v = acc[rh][t][r];
        if (mode == 0) {
          Yb[(size_t)row * N + col] = (bf16)v;
        } else if (mode == 2) {
          Yf[(size_t)row * N + col] = v;
        } else {  // mode 1: vt[b][h][d][s]
          int b = row / S, s = row - b * S;
          int h = n0 >> 6, d = 16 * t + l;
          Yb[(size_t)((b * N_HEADS + h) * HEAD_DIM + d) * S + s] = (bf16)v;
        }
      }
    }
  }
}

// ---------------------------------------------------------------------------
// RoPE applied in-place to Q and K, both laid out [B*S, D_MODEL].
// ---------------------------------------------------------------------------
__global__ __launch_bounds__(256) void rope_kernel(bf16* __restrict__ Q,
                                                   bf16* __restrict__ Kb,
                                                   int total, int S) {
  int idx = blockIdx.x * blockDim.x + threadIdx.x;
  if (idx >= total) return;
  int d = idx & 31;
  int h = (idx >> 5) & (N_HEADS - 1);
  int row = idx >> 9;
  int s = row & (S - 1);
  float inv = __powf(10000.0f, -(float)d * (1.0f / 32.0f));
  float c, sn;
  __sincosf((float)s * inv, &c, &sn);
  size_t base = (size_t)row * D_MODEL + h * HEAD_DIM + d;
  float q1 = (float)Q[base], q2 = (float)Q[base + 32];
  Q[base]      = (bf16)(q1 * c - q2 * sn);
  Q[base + 32] = (bf16)(q2 * c + q1 * sn);
  float k1 = (float)Kb[base], k2 = (float)Kb[base + 32];
  Kb[base]      = (bf16)(k1 * c - k2 * sn);
  Kb[base + 32] = (bf16)(k2 * c + k1 * sn);
}

// ---------------------------------------------------------------------------
// Flash attention (causal): 4 waves / 128 threads per block, one wave per
// 16-query tile of one (b,h).  K and V blocks staged into LDS by the Tensor
// Data Mover; S = Q K^T and O += P V on bf16 WMMA; online softmax with
// width-16 shfl row reductions; P transposed through LDS.
// Per-wave LDS: K tile 32x64 (4KB) + V tile 64x32 (4KB) + P 16x32 (1KB).
// ---------------------------------------------------------------------------
#define FA_WAVE_LDS (2048 + 2048 + 512)   // bf16 elements per wave
__global__ __launch_bounds__(128) void flash_attn_wmma(
    const bf16* __restrict__ Q, const bf16* __restrict__ Kb,
    const bf16* __restrict__ Vt, bf16* __restrict__ Ab, int S) {
  __shared__ bf16 lds[4 * FA_WAVE_LDS];

  int wave = threadIdx.x >> 5;
  int lane = threadIdx.x & 31;
  int l = lane & 15, half = lane >> 4;
  int gw = blockIdx.x * 4 + wave;
  int qtiles = S >> 4;
  int q0 = (gw % qtiles) << 4;
  int bh = gw / qtiles;
  int h = bh & (N_HEADS - 1);
  int b = bh >> 4;

  bf16* kt = lds + wave * FA_WAVE_LDS;         // [key][d]  32x64
  bf16* vtile = kt + 2048;                     // [d][key]  64x32
  bf16* pl = vtile + 2048;                     // [row][key] 16x32
  unsigned kOff = (unsigned)(wave * FA_WAVE_LDS * 2);
  unsigned vOff = kOff + 4096;

  // Q A-fragments (head-dim chunks 0..31 and 32..63)
  const bf16* qbase = Q + (size_t)(b * S + q0 + l) * D_MODEL + h * HEAD_DIM;
  v16bf aQ0 = ld_frag(qbase + half * 16,      qbase + 8 + half * 16);
  v16bf aQ1 = ld_frag(qbase + 32 + half * 16, qbase + 40 + half * 16);

  float mrow[8], lrow[8], alpha[8];
  v8f o0 = {}, o1 = {}, o2 = {}, o3 = {};
#pragma unroll
  for (int r = 0; r < 8; ++r) { mrow[r] = -1e30f; lrow[r] = 0.0f; }

  for (int jb = 0; jb <= q0; jb += 32) {
    // drain previous block's LDS reads, then stage K (32 keys x 64 dims,
    // row stride D_MODEL) and V (64 dims x 32 keys, row stride S) via TDM.
    asm volatile("s_wait_dscnt 0" ::: "memory");
    tdm_load_2d(kOff, Kb + (size_t)(b * S + jb) * D_MODEL + h * HEAD_DIM,
                64, 32, D_MODEL);
    tdm_load_2d(vOff, Vt + (size_t)((b * N_HEADS + h) * HEAD_DIM) * S + jb,
                32, 64, (unsigned)S);
    tdm_wait();

    // K B-fragments from LDS: col = key, elements = head-dims
    const bf16* kr0 = kt + (l) * 64 + half * 16;        // keys jb..+15
    const bf16* kr1 = kt + (16 + l) * 64 + half * 16;   // keys jb+16..+31
    v16bf k00 = ld_frag(kr0,      kr0 + 8);
    v16bf k01 = ld_frag(kr0 + 32, kr0 + 40);
    v16bf k10 = ld_frag(kr1,      kr1 + 8);
    v16bf k11 = ld_frag(kr1 + 32, kr1 + 40);

    v8f s0 = {}, s1 = {};
    s0 = WMMA_BF16(aQ0, k00, s0);
    s0 = WMMA_BF16(aQ1, k01, s0);
    s1 = WMMA_BF16(aQ0, k10, s1);
    s1 = WMMA_BF16(aQ1, k11, s1);

    // online softmax over this 16x32 score block (scale 1/sqrt(64))
#pragma unroll
    for (int r = 0; r < 8; ++r) {
      int rowm = q0 + r + 8 * half;
      float x0 = s0[r] * 0.125f; if (jb + l > rowm)      x0 = -1e30f;
      float x1 = s1[r] * 0.125f; if (jb + 16 + l > rowm) x1 = -1e30f;
      float v = fmaxf(x0, x1);
      v = fmaxf(v, __shfl_xor(v, 1, 16));
      v = fmaxf(v, __shfl_xor(v, 2, 16));
      v = fmaxf(v, __shfl_xor(v, 4, 16));
      v = fmaxf(v, __shfl_xor(v, 8, 16));
      float nm = fmaxf(mrow[r], v);
      float a  = __expf(mrow[r] - nm);
      float p0 = __expf(x0 - nm);
      float p1 = __expf(x1 - nm);
      float sum = p0 + p1;
      sum += __shfl_xor(sum, 1, 16);
      sum += __shfl_xor(sum, 2, 16);
      sum += __shfl_xor(sum, 4, 16);
      sum += __shfl_xor(sum, 8, 16);
      lrow[r] = lrow[r] * a + sum;
      mrow[r] = nm;
      alpha[r] = a;
      s0[r] = p0;
      s1[r] = p1;
    }
#pragma unroll
    for (int r = 0; r < 8; ++r) {
      o0[r] *= alpha[r]; o1[r] *= alpha[r]; o2[r] *= alpha[r]; o3[r] *= alpha[r];
    }

    // P: C-layout -> row-major 16x32 bf16 in LDS -> reload as A-fragment
#pragma unroll
    for (int r = 0; r < 8; ++r) {
      int rr = r + 8 * half;
      pl[rr * 32 + l]      = (bf16)s0[r];
      pl[rr * 32 + 16 + l] = (bf16)s1[r];
    }
    asm volatile("s_wait_dscnt 0" ::: "memory");
    v16bf aP = ld_frag(pl + l * 32 + half * 16, pl + l * 32 + 8 + half * 16);

    // V B-fragments from the LDS tile: col = head-dim, elements = keys
    const bf16* vr = vtile + l * 32 + half * 16;
    v16bf v0 = ld_frag(vr,            vr + 8);
    v16bf v1 = ld_frag(vr + 16 * 32,  vr + 16 * 32 + 8);
    v16bf v2 = ld_frag(vr + 32 * 32,  vr + 32 * 32 + 8);
    v16bf v3 = ld_frag(vr + 48 * 32,  vr + 48 * 32 + 8);
    o0 = WMMA_BF16(aP, v0, o0);
    o1 = WMMA_BF16(aP, v1, o1);
    o2 = WMMA_BF16(aP, v2, o2);
    o3 = WMMA_BF16(aP, v3, o3);
  }

  // normalize and store attention output [B*S, D_MODEL] bf16
#pragma unroll
  for (int r = 0; r < 8; ++r) {
    int rowm = q0 + r + 8 * half;
    float inv = 1.0f / lrow[r];
    bf16* outp = Ab + (size_t)(b * S + rowm) * D_MODEL + h * HEAD_DIM + l;
    outp[0]  = (bf16)(o0[r] * inv);
    outp[16] = (bf16)(o1[r] * inv);
    outp[32] = (bf16)(o2[r] * inv);
    outp[48] = (bf16)(o3[r] * inv);
  }
}

// ---------------------------------------------------------------------------
// Launch
// ---------------------------------------------------------------------------
extern "C" void kernel_launch(void* const* d_in, const int* in_sizes, int n_in,
                              void* d_out, int out_size, void* d_ws, size_t ws_size,
                              hipStream_t stream) {
  const float* x  = (const float*)d_in[0];
  // d_in[1] is the causal mask; causality is applied analytically.
  const float* Wq = (const float*)d_in[2];
  const float* Wk = (const float*)d_in[3];
  const float* Wv = (const float*)d_in[4];
  const float* Wo = (const float*)d_in[5];
  float* out = (float*)d_out;

  const int BS = BATCH * SEQ;  // 4096 token rows
  const int D  = D_MODEL;

  bf16* Xb  = (bf16*)d_ws;
  bf16* Wqb = Xb  + (size_t)BS * D;
  bf16* Wkb = Wqb + (size_t)D * D;
  bf16* Wvb = Wkb + (size_t)D * D;
  bf16* Wob = Wvb + (size_t)D * D;
  bf16* Qb  = Wob + (size_t)D * D;
  bf16* Kb  = Qb  + (size_t)BS * D;
  bf16* Vtb = Kb  + (size_t)BS * D;
  bf16* Ab  = Vtb + (size_t)BS * D;

  // fp32 -> bf16
  cvt_f32_bf16<<<2048, 256, 0, stream>>>(x,  Xb,  BS * D / 4);
  cvt_f32_bf16<<<1024, 256, 0, stream>>>(Wq, Wqb, D * D / 4);
  cvt_f32_bf16<<<1024, 256, 0, stream>>>(Wk, Wkb, D * D / 4);
  cvt_f32_bf16<<<1024, 256, 0, stream>>>(Wv, Wvb, D * D / 4);
  cvt_f32_bf16<<<1024, 256, 0, stream>>>(Wo, Wob, D * D / 4);

  // QKV projections: (BS/32)*(D/64) = 2048 waves = 256 blocks of 8 waves
  gemm_bf16_wmma<<<256, 256, 0, stream>>>(Xb, Wqb, Qb,  nullptr, BS, D, D, 0, SEQ);
  gemm_bf16_wmma<<<256, 256, 0, stream>>>(Xb, Wkb, Kb,  nullptr, BS, D, D, 0, SEQ);
  gemm_bf16_wmma<<<256, 256, 0, stream>>>(Xb, Wvb, Vtb, nullptr, BS, D, D, 1, SEQ);

  // RoPE on Q,K
  int rope_threads = BS * N_HEADS * 32;  // one thread per rotation pair
  rope_kernel<<<(rope_threads + 255) / 256, 256, 0, stream>>>(Qb, Kb, rope_threads, SEQ);

  // Flash attention: B*H*(S/16) = 4096 waves, 4 waves/block -> 1024 blocks
  flash_attn_wmma<<<1024, 128, 0, stream>>>(Qb, Kb, Vtb, Ab, SEQ);

  // Output projection -> fp32 d_out
  gemm_bf16_wmma<<<256, 256, 0, stream>>>(Ab, Wob, nullptr, out, BS, D, D, 2, SEQ);
}
